// Generator_13280038880015
// MI455X (gfx1250) — matrix operand order repaired and checked by reading, not compiled, not run on hardware
//
#include <hip/hip_runtime.h>
#include <hip/hip_fp16.h>

typedef __attribute__((ext_vector_type(16))) _Float16 v16h;
typedef __attribute__((ext_vector_type(8)))  _Float16 v8h;
typedef __attribute__((ext_vector_type(8)))  float    v8f;

#define THREADS 256

// ---------------- elementwise / graph-prep kernels ----------------

__global__ void k_zero(float* __restrict__ p, long long n) {
    long long i = (long long)blockIdx.x * blockDim.x + threadIdx.x;
    long long stride = (long long)gridDim.x * blockDim.x;
    for (; i < n; i += stride) p[i] = 0.0f;
}

__global__ void k_deg(const int* __restrict__ dst, float* __restrict__ deg, int E) {
    int e = blockIdx.x * blockDim.x + threadIdx.x;
    if (e < E) atomicAdd(&deg[dst[e]], 1.0f);
}

__global__ void k_dinv(float* __restrict__ deg, int N) {
    int i = blockIdx.x * blockDim.x + threadIdx.x;
    if (i < N) {
        float d = deg[i];
        deg[i] = (d > 0.0f) ? rsqrtf(fmaxf(d, 1.0f)) : 0.0f;
    }
}

__global__ void k_norm(const int* __restrict__ src, const int* __restrict__ dst,
                       const float* __restrict__ dinv, float* __restrict__ norm, int E) {
    int e = blockIdx.x * blockDim.x + threadIdx.x;
    if (e < E) norm[e] = dinv[src[e]] * dinv[dst[e]];
}

__global__ void k_concat(const float* __restrict__ cat, const float* __restrict__ noi,
                         float* __restrict__ x, int total) {
    int i = blockIdx.x * blockDim.x + threadIdx.x;
    if (i < total) {
        int node = i >> 6, f = i & 63;
        x[i] = (f < 32) ? cat[node * 32 + f] : noi[node * 32 + (f - 32)];
    }
}

// SpMM hop: hout[dst] += norm[e] * hin[src]
// One thread per (edge, 16-feature chunk): indices loaded once, float4 gathers,
// 16 native global_atomic_add_f32 (L2-resident: feature buffer = 25.6MB << 192MB L2).
__global__ void k_spmm(const int* __restrict__ src, const int* __restrict__ dst,
                       const float* __restrict__ norm, const float* __restrict__ hin,
                       float* __restrict__ hout, int E) {
    long long i = (long long)blockIdx.x * blockDim.x + threadIdx.x;
    if (i < (long long)E * 4) {
        const int e = (int)(i >> 2);
        const int c = (int)(i & 3) * 16;
        const float nrm = norm[e];
        const float4* hs = reinterpret_cast<const float4*>(hin + (long long)src[e] * 64 + c);
        float* ho = hout + (long long)dst[e] * 64 + c;
        float4 a0 = hs[0], a1 = hs[1], a2 = hs[2], a3 = hs[3];
        atomicAdd(ho + 0,  nrm * a0.x); atomicAdd(ho + 1,  nrm * a0.y);
        atomicAdd(ho + 2,  nrm * a0.z); atomicAdd(ho + 3,  nrm * a0.w);
        atomicAdd(ho + 4,  nrm * a1.x); atomicAdd(ho + 5,  nrm * a1.y);
        atomicAdd(ho + 6,  nrm * a1.z); atomicAdd(ho + 7,  nrm * a1.w);
        atomicAdd(ho + 8,  nrm * a2.x); atomicAdd(ho + 9,  nrm * a2.y);
        atomicAdd(ho + 10, nrm * a2.z); atomicAdd(ho + 11, nrm * a2.w);
        atomicAdd(ho + 12, nrm * a3.x); atomicAdd(ho + 13, nrm * a3.y);
        atomicAdd(ho + 14, nrm * a3.z); atomicAdd(ho + 15, nrm * a3.w);
    }
}

// dropout(P=0.1, hash-Bernoulli) + PReLU(0.25)
__device__ __forceinline__ float act_elem(float x, int idx, unsigned seed) {
    unsigned h = (unsigned)idx * 0x9E3779B9u + seed * 0x85EBCA6Bu;
    h ^= h >> 16; h *= 0x7feb352du; h ^= h >> 15; h *= 0x846ca68bu; h ^= h >> 16;
    x = (h >= 429496730u) ? x * (1.0f / 0.9f) : 0.0f;     // keep prob 0.9
    return (x >= 0.0f) ? x : 0.25f * x;
}

// ---------------- WMMA GEMM: Out[n,64] (+)= A[n,64] @ W[64,64] (+ bias) ----------------
// 256 threads = 8 waves; each wave owns one 16-row tile (4x 16x16 N-tiles).
// W staged once per block into LDS as f16, column-major: sW[col*64 + kin].
// B fragment (kin = 32*kc + 16*(e>>3) + 8*hh + (e&7) at fixed col) = two 16-byte LDS runs.
// If doAct: apply dropout+PReLU and store to Tgt instead of Out.
__global__ void k_gemm_wmma(const float* __restrict__ A, const float* __restrict__ W,
                            const float* __restrict__ bias, float* __restrict__ Out,
                            float* __restrict__ Tgt, int n, int addC, int doAct,
                            unsigned seed) {
    __shared__ _Float16 sW[64 * 64];   // 8 KB of the 320 KB/WGP LDS

    // cooperative W load: coalesced global reads, f32->f16 convert, col-major LDS
    for (int idx = threadIdx.x; idx < 64 * 64; idx += THREADS) {
        const int kin = idx >> 6, col = idx & 63;
        sW[col * 64 + kin] = (_Float16)W[idx];
    }
    __syncthreads();

    const int wave = threadIdx.x >> 5;
    const int lane = threadIdx.x & 31;
    const int m  = lane & 15;          // row (A) or col (B/D) within half-wave
    const int hh = lane >> 4;          // half-wave select
    const int base = (blockIdx.x * 8 + wave) * 16;

    // --- init accumulators: bias and/or existing Out (C-matrix layout, ISA 7.12.2) ---
    v8f acc[4];
#pragma unroll
    for (int nt = 0; nt < 4; ++nt) {
        const int col = nt * 16 + m;
        const float bv = bias ? bias[col] : 0.0f;
#pragma unroll
        for (int r = 0; r < 8; ++r) {
            float c = bv;
            if (addC) {
                const int row = base + r + 8 * hh;
                if (row < n) c += Out[(long long)row * 64 + col];
            }
            acc[nt][r] = c;
        }
    }

    // --- A row pointer (clamped; out-of-range rows never stored) ---
    int arow = base + m;
    if (arow >= n) arow = n - 1;
    const float* Ar = A + (long long)arow * 64;

#pragma unroll
    for (int kc = 0; kc < 2; ++kc) {   // K = 64 -> 2 steps of 32
        // A fragment: element e -> K = 32*kc + 16*(e>>3) + 8*hh + (e&7)
        v16h a;
        {
            const float4* p0 = reinterpret_cast<const float4*>(Ar + kc * 32 + hh * 8);
            const float4* p1 = reinterpret_cast<const float4*>(Ar + kc * 32 + 16 + hh * 8);
            float4 q0 = p0[0], q1 = p0[1], q2 = p1[0], q3 = p1[1];
            a[0]=(_Float16)q0.x; a[1]=(_Float16)q0.y; a[2]=(_Float16)q0.z; a[3]=(_Float16)q0.w;
            a[4]=(_Float16)q1.x; a[5]=(_Float16)q1.y; a[6]=(_Float16)q1.z; a[7]=(_Float16)q1.w;
            a[8]=(_Float16)q2.x; a[9]=(_Float16)q2.y; a[10]=(_Float16)q2.z; a[11]=(_Float16)q2.w;
            a[12]=(_Float16)q3.x; a[13]=(_Float16)q3.y; a[14]=(_Float16)q3.z; a[15]=(_Float16)q3.w;
        }
#pragma unroll
        for (int nt = 0; nt < 4; ++nt) {
            const int col = nt * 16 + m;
            // two 16B LDS vector loads: K = 32*kc + 8*hh + e and 32*kc + 16 + 8*hh + e
            const v8h* wp = reinterpret_cast<const v8h*>(&sW[col * 64 + kc * 32 + hh * 8]);
            v8h b0 = wp[0], b1 = wp[2];
            v16h b;
#pragma unroll
            for (int e = 0; e < 8; ++e) { b[e] = b0[e]; b[8 + e] = b1[e]; }
            acc[nt] = __builtin_amdgcn_wmma_f32_16x16x32_f16(
                false, a, false, b, (short)0, acc[nt], false, false);
        }
    }

    // --- store D (predicated; after WMMA so EXEC is all-1s during WMMA) ---
#pragma unroll
    for (int nt = 0; nt < 4; ++nt) {
        const int col = nt * 16 + m;
#pragma unroll
        for (int r = 0; r < 8; ++r) {
            const int row = base + r + 8 * hh;
            if (row < n) {
                const long long idx = (long long)row * 64 + col;
                if (doAct) Tgt[idx] = act_elem(acc[nt][r], (int)idx, seed);
                else       Out[idx] = acc[nt][r];
            }
        }
    }
}

// ---------------- host-side orchestration ----------------

static inline unsigned cdiv(long long a, long long b) { return (unsigned)((a + b - 1) / b); }

extern "C" void kernel_launch(void* const* d_in, const int* in_sizes, int n_in,
                              void* d_out, int out_size, void* d_ws, size_t ws_size,
                              hipStream_t stream) {
    const float* category = (const float*)d_in[0];
    const float* noise    = (const float*)d_in[1];
    const int*   edge     = (const int*)d_in[2];
    const float* Wl[3] = { (const float*)d_in[3], (const float*)d_in[5], (const float*)d_in[7] };
    const float* Bl[3] = { (const float*)d_in[4], (const float*)d_in[6], (const float*)d_in[8] };

    const int N = in_sizes[0] / 32;
    const int E = in_sizes[2] / 2;
    const int* src = edge;
    const int* dst = edge + E;

    // workspace layout
    char* ws = (char*)d_ws;
    size_t off = 0;
    auto carve = [&](size_t bytes) -> char* {
        char* p = ws + off;
        off = (off + bytes + 255) & ~(size_t)255;
        return p;
    };
    const size_t featBytes = (size_t)N * 64 * sizeof(float);
    float* deg  = (float*)carve((size_t)N * sizeof(float));
    float* norm = (float*)carve((size_t)E * sizeof(float));
    float* x    = (float*)carve(featBytes);
    float* h0   = (float*)carve(featBytes);
    float* h1   = (float*)carve(featBytes);
    float* outb = (float*)carve(featBytes);

    const long long feat = (long long)N * 64;

    // --- graph normalization ---
    k_zero<<<cdiv(N, THREADS), THREADS, 0, stream>>>(deg, N);
    k_deg<<<cdiv(E, THREADS), THREADS, 0, stream>>>(dst, deg, E);
    k_dinv<<<cdiv(N, THREADS), THREADS, 0, stream>>>(deg, N);
    k_norm<<<cdiv(E, THREADS), THREADS, 0, stream>>>(src, dst, deg, norm, E);

    // --- x = concat(category, noise) ---
    k_concat<<<cdiv(feat, THREADS), THREADS, 0, stream>>>(category, noise, x, (int)feat);

    const unsigned gemmBlocks = cdiv((N + 15) / 16, 8);
    const unsigned spmmBlocks = cdiv((long long)E * 4, THREADS);
    float* hb[2] = { h0, h1 };

    for (int layer = 0; layer < 3; ++layer) {
        // out = x @ W[0] + b
        k_gemm_wmma<<<gemmBlocks, THREADS, 0, stream>>>(
            x, Wl[layer], Bl[layer], outb, nullptr, N, 0, 0, 0u);

        const float* hprev = x;
        for (int k = 1; k <= 3; ++k) {
            float* hc = hb[(k - 1) & 1];
            k_zero<<<cdiv(feat, THREADS), THREADS, 0, stream>>>(hc, feat);
            k_spmm<<<spmmBlocks, THREADS, 0, stream>>>(src, dst, norm, hprev, hc, E);
            const int last = (k == 3);
            float* target = last ? ((layer == 2) ? (float*)d_out : x) : nullptr;
            // final hop: fuse dropout+PReLU into the accumulate-GEMM store
            k_gemm_wmma<<<gemmBlocks, THREADS, 0, stream>>>(
                hc, Wl[layer] + (size_t)k * 64 * 64, nullptr, outb, target,
                N, 1, last, (unsigned)layer);
            hprev = hc;
        }
    }
    (void)n_in; (void)out_size; (void)ws_size;
}
